// Attention_61357902791110
// MI455X (gfx1250) — matrix operand order repaired and checked by reading, not compiled, not run on hardware
//
#include <hip/hip_runtime.h>
#include <hip/hip_bf16.h>

typedef __attribute__((ext_vector_type(16))) __bf16 v16bf;
typedef __attribute__((ext_vector_type(8)))  __bf16 v8bf;
typedef __attribute__((ext_vector_type(8)))  float  v8f;

#define B_ 128
#define P_ 196
#define E_ 2048
#define D_ 512
#define A_ 512
#define M_TOTAL (B_ * P_)   // 25088 = 784 * 32

// ---------------------------------------------------------------------------
// Kernel 0a: transpose + convert We [E,A] fp32 -> WeT [A,E] bf16 (LDS tiled)
// ---------------------------------------------------------------------------
__global__ __launch_bounds__(256) void k_transpose_we(
    const float* __restrict__ We, __bf16* __restrict__ WeT)
{
  __shared__ float tile[32][33];
  const int a0 = blockIdx.x * 32;   // A/32 = 16
  const int e0 = blockIdx.y * 32;   // E/32 = 64
  const int tx = threadIdx.x & 31;
  const int ty = threadIdx.x >> 5;  // 0..7
#pragma unroll
  for (int i = 0; i < 4; ++i)
    tile[ty + 8 * i][tx] = We[(size_t)(e0 + ty + 8 * i) * A_ + (a0 + tx)];
  __syncthreads();
#pragma unroll
  for (int i = 0; i < 4; ++i)
    WeT[(size_t)(a0 + ty + 8 * i) * E_ + (e0 + tx)] = (__bf16)tile[tx][ty + 8 * i];
}

// ---------------------------------------------------------------------------
// Kernel 0b: comb[b][a] = dot(dh[b,:], Wd[:,a]) + bd[a] + be[a]   (tiny GEMM)
// ---------------------------------------------------------------------------
__global__ __launch_bounds__(256) void k_comb(
    const float* __restrict__ dh, const float* __restrict__ Wd,
    const float* __restrict__ bd, const float* __restrict__ be,
    float* __restrict__ comb)
{
  const int b = blockIdx.x;
  const int a = threadIdx.x * 2;
  float acc0 = bd[a] + be[a];
  float acc1 = bd[a + 1] + be[a + 1];
  const float* dhb = dh + (size_t)b * D_;
  for (int d = 0; d < D_; ++d) {
    const float x = dhb[d];
    const float2 w = *(const float2*)(Wd + (size_t)d * A_ + a);
    acc0 = fmaf(x, w.x, acc0);
    acc1 = fmaf(x, w.y, acc1);
  }
  comb[(size_t)b * A_ + a]     = acc0;
  comb[(size_t)b * A_ + a + 1] = acc1;
}

// ---------------------------------------------------------------------------
// Kernel 1: fused scores. att[row] = sum_a relu(enc@We + comb)[row,a]*Wf[a]+bf
// Block = 32 rows x all 512 cols (enc read once). 8 waves, wave w owns cols
// [64w, 64w+64): 2 M-tiles x 4 N-tiles of v_wmma_f32_16x16x32_bf16.
// ---------------------------------------------------------------------------
__device__ __forceinline__ v16bf load_a_frag(const float* __restrict__ s) {
  // A operand in this code's K order (B matches): lane-half h holds contiguous K 16h..16h+15 in v0-7
  const float4 f0 = *(const float4*)(s);
  const float4 f1 = *(const float4*)(s + 4);
  const float4 f2 = *(const float4*)(s + 8);
  const float4 f3 = *(const float4*)(s + 12);
  v16bf r;
  r[0]  = (__bf16)f0.x; r[1]  = (__bf16)f0.y; r[2]  = (__bf16)f0.z; r[3]  = (__bf16)f0.w;
  r[4]  = (__bf16)f1.x; r[5]  = (__bf16)f1.y; r[6]  = (__bf16)f1.z; r[7]  = (__bf16)f1.w;
  r[8]  = (__bf16)f2.x; r[9]  = (__bf16)f2.y; r[10] = (__bf16)f2.z; r[11] = (__bf16)f2.w;
  r[12] = (__bf16)f3.x; r[13] = (__bf16)f3.y; r[14] = (__bf16)f3.z; r[15] = (__bf16)f3.w;
  return r;
}

__global__ __launch_bounds__(256) void k_scores(
    const float* __restrict__ enc, const __bf16* __restrict__ WeT,
    const float* __restrict__ comb, const float* __restrict__ Wf,
    const float* __restrict__ bfp, float* __restrict__ att)
{
  __shared__ float sc[8][32];
  const int t     = threadIdx.x;
  const int lane  = t & 31;
  const int wave  = t >> 5;
  const int lo16  = lane & 15;
  const int hisel = lane >> 4;
  const int row0  = blockIdx.x * 32;

  v8f c[2][4];
  const v8f vz = {0.f, 0.f, 0.f, 0.f, 0.f, 0.f, 0.f, 0.f};
#pragma unroll
  for (int mt = 0; mt < 2; ++mt)
#pragma unroll
    for (int nt = 0; nt < 4; ++nt) c[mt][nt] = vz;

  const float* arow[2];
  arow[0] = enc + (size_t)(row0 + lo16) * E_;
  arow[1] = arow[0] + (size_t)16 * E_;
  const __bf16* brow[4];
#pragma unroll
  for (int nt = 0; nt < 4; ++nt)
    brow[nt] = WeT + (size_t)(wave * 64 + nt * 16 + lo16) * E_;

  for (int k0 = 0; k0 < E_; k0 += 32) {
    const int ka = k0 + hisel * 16;   // A operand: 16 contiguous K per half
    const int kb = k0 + hisel * 16;   // B operand: 16 contiguous K per half
    v16bf afrag[2], bfrag[4];
#pragma unroll
    for (int mt = 0; mt < 2; ++mt) afrag[mt] = load_a_frag(arow[mt] + ka);
#pragma unroll
    for (int nt = 0; nt < 4; ++nt) bfrag[nt] = *(const v16bf*)(brow[nt] + kb);
#pragma unroll
    for (int mt = 0; mt < 2; ++mt)
#pragma unroll
      for (int nt = 0; nt < 4; ++nt)
        c[mt][nt] = __builtin_amdgcn_wmma_f32_16x16x32_bf16(
            false, afrag[mt], false, bfrag[nt], (short)0, c[mt][nt], false, false);
  }

  // Epilogue: relu(c + comb) * Wf, reduce over this wave's 64 columns.
  // C layout: VGPR j -> M=j (lanes 0-15), M=j+8 (lanes 16-31); N = lane%16.
#pragma unroll
  for (int mt = 0; mt < 2; ++mt) {
#pragma unroll
    for (int j = 0; j < 8; ++j) {
      const int rl   = mt * 16 + hisel * 8 + j;
      const int grow = row0 + rl;
      const int bb   = grow / P_;
      float s = 0.0f;
#pragma unroll
      for (int nt = 0; nt < 4; ++nt) {
        const int a = wave * 64 + nt * 16 + lo16;
        float v = c[mt][nt][j] + comb[(size_t)bb * A_ + a];
        v = fmaxf(v, 0.0f);
        s = fmaf(v, Wf[a], s);
      }
      s += __shfl_xor(s, 8, 32);
      s += __shfl_xor(s, 4, 32);
      s += __shfl_xor(s, 2, 32);
      s += __shfl_xor(s, 1, 32);
      if (lo16 == 0) sc[wave][rl] = s;  // deterministic: no atomics
    }
  }
  __syncthreads();
  if (t < 32) {
    float s = 0.0f;
#pragma unroll
    for (int w = 0; w < 8; ++w) s += sc[w][t];
    att[row0 + t] = s + bfp[0];
  }
}

// ---------------------------------------------------------------------------
// Kernel 2: per-batch softmax over P, then awe[b,e] = sum_p alpha[p]*enc[b,p,e]
// ---------------------------------------------------------------------------
__global__ __launch_bounds__(256) void k_softmax_awe(
    const float* __restrict__ enc, const float* __restrict__ att,
    float* __restrict__ alpha, float* __restrict__ awe)
{
  __shared__ float satt[P_];
  __shared__ float wmax[8];
  __shared__ float wsum[8];
  const int b    = blockIdx.x;
  const int t    = threadIdx.x;
  const int lane = t & 31;
  const int wv   = t >> 5;

  const float v = (t < P_) ? att[(size_t)b * P_ + t] : -3.4e38f;
  float m = v;
#pragma unroll
  for (int off = 16; off; off >>= 1) m = fmaxf(m, __shfl_xor(m, off, 32));
  if (lane == 0) wmax[wv] = m;
  __syncthreads();
  float gm = wmax[0];
#pragma unroll
  for (int i = 1; i < 8; ++i) gm = fmaxf(gm, wmax[i]);

  const float e = (t < P_) ? __expf(v - gm) : 0.0f;
  float s = e;
#pragma unroll
  for (int off = 16; off; off >>= 1) s += __shfl_xor(s, off, 32);
  if (lane == 0) wsum[wv] = s;
  __syncthreads();
  float gs = 0.0f;
#pragma unroll
  for (int i = 0; i < 8; ++i) gs += wsum[i];
  const float inv = 1.0f / gs;
  if (t < P_) {
    const float al = e * inv;
    satt[t] = al;
    alpha[(size_t)b * P_ + t] = al;
  }
  __syncthreads();

  // each thread owns 8 contiguous e-values -> fully coalesced float4 streams
  float4 a0 = {0.f, 0.f, 0.f, 0.f}, a1 = {0.f, 0.f, 0.f, 0.f};
  const float* encb = enc + (size_t)b * P_ * E_ + t * 8;
  for (int p = 0; p < P_; ++p) {
    const float ap = satt[p];
    const float4 x0 = *(const float4*)(encb + (size_t)p * E_);
    const float4 x1 = *(const float4*)(encb + (size_t)p * E_ + 4);
    a0.x = fmaf(ap, x0.x, a0.x); a0.y = fmaf(ap, x0.y, a0.y);
    a0.z = fmaf(ap, x0.z, a0.z); a0.w = fmaf(ap, x0.w, a0.w);
    a1.x = fmaf(ap, x1.x, a1.x); a1.y = fmaf(ap, x1.y, a1.y);
    a1.z = fmaf(ap, x1.z, a1.z); a1.w = fmaf(ap, x1.w, a1.w);
  }
  float* o = awe + (size_t)b * E_ + t * 8;
  *(float4*)(o)     = a0;
  *(float4*)(o + 4) = a1;
}

// ---------------------------------------------------------------------------
extern "C" void kernel_launch(void* const* d_in, const int* in_sizes, int n_in,
                              void* d_out, int out_size, void* d_ws, size_t ws_size,
                              hipStream_t stream) {
  (void)in_sizes; (void)n_in; (void)out_size; (void)ws_size;
  const float* enc = (const float*)d_in[0];   // [B,P,E]
  const float* dh  = (const float*)d_in[1];   // [B,D]
  const float* We  = (const float*)d_in[2];   // [E,A]
  const float* be  = (const float*)d_in[3];   // [A]
  const float* Wd  = (const float*)d_in[4];   // [D,A]
  const float* bd  = (const float*)d_in[5];   // [A]
  const float* Wf  = (const float*)d_in[6];   // [A,1]
  const float* bfp = (const float*)d_in[7];   // [1]

  float* out   = (float*)d_out;
  float* awe   = out;                 // [B,E] = 262144 floats
  float* alpha = out + B_ * E_;       // [B,P] = 25088 floats

  char*   ws   = (char*)d_ws;
  __bf16* WeT  = (__bf16*)ws;                                   // 2 MB
  float*  comb = (float*)(ws + (size_t)A_ * E_ * 2);            // 256 KB
  float*  att  = (float*)(ws + (size_t)A_ * E_ * 2 + (size_t)B_ * A_ * 4); // 100 KB

  k_transpose_we<<<dim3(A_ / 32, E_ / 32), 256, 0, stream>>>(We, WeT);
  k_comb<<<B_, 256, 0, stream>>>(dh, Wd, bd, be, comb);
  k_scores<<<M_TOTAL / 32, 256, 0, stream>>>(enc, WeT, comb, Wf, bfp, att);
  k_softmax_awe<<<B_, 256, 0, stream>>>(enc, att, alpha, awe);
}
